// UIElementDetector_77833397338470
// MI455X (gfx1250) — compile-verified, matrix operand order checked by reading, not run on hardware
//
#include <hip/hip_runtime.h>
#include <hip/hip_bf16.h>
#include <stdint.h>

// ---------------------------------------------------------------------------
// Types for CDNA5 WMMA (wave32): v_wmma_f32_16x16x32_f16
// ---------------------------------------------------------------------------
typedef __attribute__((ext_vector_type(16))) _Float16 v16h;
typedef __attribute__((ext_vector_type(8)))  _Float16 v8h;
typedef __attribute__((ext_vector_type(8)))  float    v8f;

#define HH 768
#define WW 768
#define IW 769               // integral image dim
#define C1C 32
#define C2C 64
#define NBOX 512
#define KPOOL 1600           // C2C * 5 * 5
#define BN_EPS 1e-5f

// A/B fragment loader for 16x16x32 f16 WMMA.
// ISA layout (16-bit A 16x32): lane l holds row (l&15); lanes 0-15 carry
// K in [0,8)+[16,24), lanes 16-31 carry K in [8,16)+[24,32).  With data laid
// out K-contiguous, lane l reads two 16-byte chunks at k0=(l>>4)*8 and k0+16.
// `p` must already point at element k0 of the 32-long K vector.
__device__ __forceinline__ v16h load_frag16(const _Float16* __restrict__ p, bool valid) {
  v16h f;
  if (valid) {
    v8h lo = *(const v8h*)(p);
    v8h hi = *(const v8h*)(p + 16);
#pragma unroll
    for (int e = 0; e < 8; ++e) { f[e] = lo[e]; f[e + 8] = hi[e]; }
  } else {
#pragma unroll
    for (int e = 0; e < 16; ++e) f[e] = (_Float16)0.0f;
  }
  return f;
}

// Lane broadcast / gather via LDS-permute hardware (no LDS memory touched).
__device__ __forceinline__ float lane_gather(float v, int src_lane) {
  int r = __builtin_amdgcn_ds_bpermute(src_lane << 2, __builtin_bit_cast(int, v));
  return __builtin_bit_cast(float, r);
}

// Inclusive Hillis-Steele scan across a wave32.
__device__ __forceinline__ float wave_incl_scan(float v, int lane) {
#pragma unroll
  for (int off = 1; off < 32; off <<= 1) {
    float n = lane_gather(v, lane - off);   // wraps for lane<off; masked below
    if (lane >= off) v += n;
  }
  return v;
}

// ---------------------------------------------------------------------------
// Weight pre-pack kernels (fp32 -> f16, K-contiguous blocks of 32)
// ---------------------------------------------------------------------------
// conv2_w (64,32,3,3) -> wp2[s][n][ci], s = dy*3+dx
__global__ void pack_w2_kernel(const float* __restrict__ w, _Float16* __restrict__ wp) {
  int idx = blockIdx.x * 256 + threadIdx.x;
  if (idx >= 9 * 64 * 32) return;
  int s  = idx / 2048;
  int r  = idx % 2048;
  int n  = r / 32;
  int ci = r % 32;
  wp[idx] = (_Float16)w[(n * 32 + ci) * 9 + s];
}

// fc1_w (128,1600) -> wpf[ks][n][kc], ks in [0,50)
__global__ void pack_fc1_kernel(const float* __restrict__ w, _Float16* __restrict__ wp) {
  int idx = blockIdx.x * 256 + threadIdx.x;
  if (idx >= 50 * 128 * 32) return;
  int ks = idx / 4096;
  int r  = idx % 4096;
  int n  = r / 32;
  int kc = r % 32;
  wp[idx] = (_Float16)w[n * 1600 + ks * 32 + kc];
}

// ---------------------------------------------------------------------------
// conv1 (3->32, 3x3, pad 1) + bias + ReLU + BN1, direct VALU conv.
// Output x1 in pixel-major [H][W][32] f16 (so conv2 A-fragments are contiguous)
// ---------------------------------------------------------------------------
__global__ __launch_bounds__(256) void conv1_kernel(
    const float* __restrict__ img, const float* __restrict__ w1,
    const float* __restrict__ b1, const float* __restrict__ g1,
    const float* __restrict__ beta1, const float* __restrict__ m1,
    const float* __restrict__ v1, _Float16* __restrict__ x1) {
  __shared__ float wl[C1C * 27];
  __shared__ float sc[C1C], sh[C1C], bb[C1C];
  for (int i = threadIdx.x; i < C1C * 27; i += 256) wl[i] = w1[i];
  if (threadIdx.x < C1C) {
    int c = threadIdx.x;
    bb[c] = b1[c];
    float s = g1[c] * rsqrtf(v1[c] + BN_EPS);
    sc[c] = s;
    sh[c] = beta1[c] - m1[c] * s;
  }
  __syncthreads();

  int pix = blockIdx.x * 256 + threadIdx.x;   // grid sized exactly 768*768
  int y = pix / WW, x = pix % WW;
  float acc[C1C];
#pragma unroll
  for (int c = 0; c < C1C; ++c) acc[c] = 0.0f;

  for (int ci = 0; ci < 3; ++ci)
    for (int dy = 0; dy < 3; ++dy) {
      int py = y + dy - 1;
      for (int dx = 0; dx < 3; ++dx) {
        int px = x + dx - 1;
        float v = ((unsigned)py < (unsigned)HH && (unsigned)px < (unsigned)WW)
                      ? img[(ci * HH + py) * WW + px] : 0.0f;
        int t = ci * 9 + dy * 3 + dx;
#pragma unroll
        for (int c = 0; c < C1C; ++c) acc[c] = fmaf(v, wl[c * 27 + t], acc[c]);
      }
    }

  _Float16* o = x1 + ((size_t)pix << 5);
#pragma unroll
  for (int c = 0; c < C1C; ++c) {
    float val = fmaxf(acc[c] + bb[c], 0.0f) * sc[c] + sh[c];
    o[c] = (_Float16)val;
  }
}

// ---------------------------------------------------------------------------
// Zero row-0 / col-0 of the 64 integral-image planes
// ---------------------------------------------------------------------------
__global__ void zero_border_kernel(float* __restrict__ integ) {
  int idx = blockIdx.x * 256 + threadIdx.x;
  if (idx >= C2C * 2 * IW) return;
  int c = idx / (2 * IW);
  int r = idx % (2 * IW);
  float* I = integ + (size_t)c * IW * IW;
  if (r < IW) I[r] = 0.0f;                    // row 0
  else        I[(size_t)(r - IW) * IW] = 0.0f; // col 0
}

// ---------------------------------------------------------------------------
// conv2 (32->64, 3x3, pad 1) + bias + ReLU + BN2 as implicit GEMM via WMMA.
// One wave per (16-pixel strip x 64 channels) tile.  K = 9 taps x 32 chans.
// Writes the BN'd feature directly into the interior of the integral planes
// (planar [C][769][769]) via an LDS transpose for coalesced stores.
// ---------------------------------------------------------------------------
__global__ __launch_bounds__(32) void conv2_wmma_kernel(
    const _Float16* __restrict__ x1, const _Float16* __restrict__ wp2,
    const float* __restrict__ b2, const float* __restrict__ g2,
    const float* __restrict__ beta2, const float* __restrict__ m2,
    const float* __restrict__ v2, float* __restrict__ integ) {
  const int x0   = blockIdx.x * 16;
  const int y    = blockIdx.y;
  const int lane = threadIdx.x;
  const int hf   = lane >> 4;   // 0/1 : K-octet select
  const int mn   = lane & 15;   // row (A) / col (B,D) index

  v8f zero = {};
  v8f acc[4] = {zero, zero, zero, zero};

#pragma unroll
  for (int s = 0; s < 9; ++s) {
    const int dy = s / 3, dx = s % 3;
    const int py = y + dy - 1;
    const int px = x0 + mn + dx - 1;
    const bool valid = ((unsigned)py < (unsigned)HH) && ((unsigned)px < (unsigned)WW);
    const _Float16* pa = x1 + (((size_t)py * WW + px) << 5) + hf * 8;
    v16h a = load_frag16(pa, valid);
#pragma unroll
    for (int nt = 0; nt < 4; ++nt) {
      const _Float16* pb = wp2 + (((size_t)(s * 64 + nt * 16 + mn)) << 5) + hf * 8;
      v16h b = load_frag16(pb, true);
      acc[nt] = __builtin_amdgcn_wmma_f32_16x16x32_f16(
          false, a, false, b, (short)0, acc[nt], false, false);
    }
  }

  // Epilogue: bias + ReLU + BN, stage 64x16 tile in LDS, coalesced store.
  __shared__ float tile[C2C][17];
#pragma unroll
  for (int nt = 0; nt < 4; ++nt) {
    const int n = nt * 16 + mn;
    const float bias = b2[n];
    const float sc   = g2[n] * rsqrtf(v2[n] + BN_EPS);
    const float sh   = beta2[n] - m2[n] * sc;
#pragma unroll
    for (int r = 0; r < 8; ++r) {
      const int m = r + hf * 8;                 // pixel column within strip
      float val = fmaxf(acc[nt][r] + bias, 0.0f) * sc + sh;
      tile[n][m] = val;
    }
  }
  __syncthreads();
  // 64 rows x 16 cols = 1024 floats; 32 lanes, consecutive lanes -> consecutive x
#pragma unroll
  for (int k = 0; k < 32; ++k) {
    int t   = k * 32 + lane;
    int row = t >> 4;        // channel
    int col = t & 15;        // pixel within strip
    integ[((size_t)row * IW + (y + 1)) * IW + (x0 + col + 1)] = tile[row][col];
  }
}

// ---------------------------------------------------------------------------
// Integral image pass 1: cumsum over y.  Thread per (c,x) column: adjacent
// lanes touch adjacent x -> every access is a coalesced 128B burst per wave-row.
// ---------------------------------------------------------------------------
__global__ void cumsum_y_kernel(float* __restrict__ integ) {
  int idx = blockIdx.x * 256 + threadIdx.x;
  if (idx >= C2C * WW) return;
  int c = idx / WW;
  int x = idx % WW + 1;
  float* I = integ + (size_t)c * IW * IW;
  float s = 0.0f;
  for (int y = 1; y <= HH; ++y) {
    s += I[(size_t)y * IW + x];
    I[(size_t)y * IW + x] = s;
  }
}

// ---------------------------------------------------------------------------
// Integral image pass 2: cumsum over x as a WAVE-PER-ROW scan.
// 32 consecutive floats per load (coalesced), inclusive wave scan via
// ds_bpermute lane shuffles, carry chained across the 24 segments of a row.
// 8 waves (8 rows) per 256-thread block.
// ---------------------------------------------------------------------------
__global__ __launch_bounds__(256) void cumsum_x_kernel(float* __restrict__ integ) {
  const int lane = threadIdx.x & 31;
  const int row_id = blockIdx.x * 8 + (threadIdx.x >> 5);   // 0 .. 64*768-1
  const int c = row_id / HH;
  const int y = row_id % HH + 1;
  float* row = integ + (size_t)c * IW * IW + (size_t)y * IW;

  float carry = 0.0f;
#pragma unroll 1
  for (int seg = 0; seg < WW / 32; ++seg) {      // 24 segments of 32
    const int x = 1 + seg * 32 + lane;
    float v = row[x];
    v = wave_incl_scan(v, lane);
    v += carry;
    row[x] = v;
    carry = lane_gather(v, 31);                  // broadcast row total so far
  }
}

// ---------------------------------------------------------------------------
// ROI average pool via 4 integral lookups -> flat [512][1600] f16
// flat[b, c*25 + i*5 + j]
// ---------------------------------------------------------------------------
__global__ void roipool_kernel(const float* __restrict__ integ,
                               const int* __restrict__ boxes,
                               _Float16* __restrict__ flat) {
  int idx = blockIdx.x * 256 + threadIdx.x;
  if (idx >= NBOX * KPOOL) return;
  int b  = idx / KPOOL;
  int r  = idx % KPOOL;
  int c  = r / 25;
  int ij = r % 25;
  int i = ij / 5, j = ij % 5;
  int xmin = boxes[b * 4 + 0], ymin = boxes[b * 4 + 1];
  int xmax = boxes[b * 4 + 2], ymax = boxes[b * 4 + 3];
  int bh = ymax - ymin, bw = xmax - xmin;
  int y0 = ymin + (i * bh) / 5;
  int y1 = ymin + ((i + 1) * bh + 4) / 5;
  int x0 = xmin + (j * bw) / 5;
  int x1 = xmin + ((j + 1) * bw + 4) / 5;
  const float* I = integ + (size_t)c * IW * IW;
  float s = I[(size_t)y1 * IW + x1] - I[(size_t)y0 * IW + x1]
          - I[(size_t)y1 * IW + x0] + I[(size_t)y0 * IW + x0];
  float area = (float)((y1 - y0) * (x1 - x0));
  flat[idx] = (_Float16)(s / area);
}

// ---------------------------------------------------------------------------
// FC1: (512 x 1600) @ (1600 x 128) + bias, ReLU -> h1 f16, via WMMA.
// One wave per 16x16 output tile; K = 50 steps of 32.
// ---------------------------------------------------------------------------
__global__ __launch_bounds__(32) void fc1_wmma_kernel(
    const _Float16* __restrict__ flat, const _Float16* __restrict__ wp,
    const float* __restrict__ fb, _Float16* __restrict__ h1) {
  const int mt = blockIdx.x;   // 0..31
  const int nt = blockIdx.y;   // 0..7
  const int lane = threadIdx.x;
  const int hf = lane >> 4;
  const int mn = lane & 15;

  v8f acc = {};
  for (int ks = 0; ks < 50; ++ks) {
    const _Float16* pa = flat + (size_t)(mt * 16 + mn) * KPOOL + ks * 32 + hf * 8;
    const _Float16* pb = wp + ((size_t)(ks * 128 + nt * 16 + mn)) * 32 + hf * 8;
    v16h a = load_frag16(pa, true);
    v16h b = load_frag16(pb, true);
    acc = __builtin_amdgcn_wmma_f32_16x16x32_f16(
        false, a, false, b, (short)0, acc, false, false);
  }
  const int n = nt * 16 + mn;
  const float bias = fb[n];
#pragma unroll
  for (int r = 0; r < 8; ++r) {
    const int m = mt * 16 + r + hf * 8;
    float v = fmaxf(acc[r] + bias, 0.0f);
    h1[(size_t)m * 128 + n] = (_Float16)v;
  }
}

// ---------------------------------------------------------------------------
// FC2: (512 x 128) @ (128 x 4) + bias -> fp32 output. Tiny; VALU.
// ---------------------------------------------------------------------------
__global__ void fc2_kernel(const _Float16* __restrict__ h1,
                           const float* __restrict__ w,
                           const float* __restrict__ b,
                           float* __restrict__ out) {
  int idx = blockIdx.x * 256 + threadIdx.x;
  if (idx >= NBOX * 4) return;
  int m = idx / 4, cls = idx % 4;
  float s = b[cls];
  for (int k = 0; k < 128; ++k)
    s += (float)h1[(size_t)m * 128 + k] * w[cls * 128 + k];
  out[idx] = s;
}

// ---------------------------------------------------------------------------
// Launcher
// ---------------------------------------------------------------------------
extern "C" void kernel_launch(void* const* d_in, const int* in_sizes, int n_in,
                              void* d_out, int out_size, void* d_ws, size_t ws_size,
                              hipStream_t stream) {
  (void)in_sizes; (void)n_in; (void)out_size; (void)ws_size;
  const float* image   = (const float*)d_in[0];
  const int*   boxes   = (const int*)d_in[1];
  const float* conv1_w = (const float*)d_in[2];
  const float* conv1_b = (const float*)d_in[3];
  const float* bn1_g   = (const float*)d_in[4];
  const float* bn1_b   = (const float*)d_in[5];
  const float* bn1_m   = (const float*)d_in[6];
  const float* bn1_v   = (const float*)d_in[7];
  const float* conv2_w = (const float*)d_in[8];
  const float* conv2_b = (const float*)d_in[9];
  const float* bn2_g   = (const float*)d_in[10];
  const float* bn2_b   = (const float*)d_in[11];
  const float* bn2_m   = (const float*)d_in[12];
  const float* bn2_v   = (const float*)d_in[13];
  const float* fc1_w   = (const float*)d_in[14];
  const float* fc1_b   = (const float*)d_in[15];
  const float* fc2_w   = (const float*)d_in[16];
  const float* fc2_b   = (const float*)d_in[17];
  float* out = (float*)d_out;

  // Workspace carve-up (256B aligned)
  char* ws = (char*)d_ws;
  size_t off = 0;
  auto carve = [&](size_t bytes) -> void* {
    void* p = ws + off;
    off = (off + bytes + 255) & ~(size_t)255;
    return p;
  };
  _Float16* x1    = (_Float16*)carve((size_t)HH * WW * C1C * 2);        // 37.7 MB
  float*    integ = (float*)carve((size_t)C2C * IW * IW * 4);           // 151 MB
  _Float16* wp2   = (_Float16*)carve((size_t)9 * 64 * 32 * 2);
  _Float16* flat  = (_Float16*)carve((size_t)NBOX * KPOOL * 2);
  _Float16* wpf   = (_Float16*)carve((size_t)50 * 128 * 32 * 2);
  _Float16* h1    = (_Float16*)carve((size_t)NBOX * 128 * 2);

  // 1. pack weights
  pack_w2_kernel<<<(9 * 64 * 32 + 255) / 256, 256, 0, stream>>>(conv2_w, wp2);
  pack_fc1_kernel<<<(50 * 128 * 32 + 255) / 256, 256, 0, stream>>>(fc1_w, wpf);

  // 2. conv1 + ReLU + BN1 -> x1 (f16, pixel-major)
  conv1_kernel<<<(HH * WW) / 256, 256, 0, stream>>>(
      image, conv1_w, conv1_b, bn1_g, bn1_b, bn1_m, bn1_v, x1);

  // 3. zero integral borders, conv2 (WMMA) writes interior
  zero_border_kernel<<<(C2C * 2 * IW + 255) / 256, 256, 0, stream>>>(integ);
  conv2_wmma_kernel<<<dim3(WW / 16, HH), 32, 0, stream>>>(
      x1, wp2, conv2_b, bn2_g, bn2_b, bn2_m, bn2_v, integ);

  // 4. integral image: coalesced y-pass, then wave-scan x-pass
  cumsum_y_kernel<<<(C2C * WW + 255) / 256, 256, 0, stream>>>(integ);
  cumsum_x_kernel<<<(C2C * HH) / 8, 256, 0, stream>>>(integ);

  // 5. ROI pool -> flat f16
  roipool_kernel<<<(NBOX * KPOOL + 255) / 256, 256, 0, stream>>>(integ, boxes, flat);

  // 6. FC1 (WMMA) -> h1, FC2 -> out
  fc1_wmma_kernel<<<dim3(NBOX / 16, 128 / 16), 32, 0, stream>>>(flat, wpf, fc1_b, h1);
  fc2_kernel<<<(NBOX * 4 + 255) / 256, 256, 0, stream>>>(h1, fc2_w, fc2_b, out);
}
